// TransformerDecoderLayer_26895085207757
// MI455X (gfx1250) — matrix operand order repaired and checked by reading, not compiled, not run on hardware
//
#include <hip/hip_runtime.h>
#include <hip/hip_bf16.h>

// ---------------------------------------------------------------------------
// Transformer decoder layer on gfx1250 (MI455X), wave32 + WMMA f16.
//   B=4, S=2048, d=1024, f=4096, topk = 1024.
// GEMMs: V_WMMA_F32_16X16X32_F16, double-buffered LDS tiles fed by
// GLOBAL_LOAD_ASYNC_TO_LDS_B128 (ASYNCcnt) when available.
// ---------------------------------------------------------------------------

typedef __attribute__((ext_vector_type(16))) _Float16 v16h;
typedef __attribute__((ext_vector_type(8)))  _Float16 v8h;
typedef __attribute__((ext_vector_type(8)))  float    v8f;
typedef __attribute__((ext_vector_type(4)))  int      v4i;

#define BATCH 4
#define SEQ   2048
#define DMODEL 1024
#define DFF   4096
#define BS    (BATCH * SEQ)       // 8192 rows
#define TOPK  1024                // SPARSITY * SEQ
#define LN_EPS 1e-6f
#define TKS   64                  // K-slab per pipeline stage (2 WMMA K-tiles)

#if defined(__gfx1250__) && __has_builtin(__builtin_amdgcn_global_load_async_to_lds_b128)
#define ASYNC_LDS 1
#else
#define ASYNC_LDS 0
#endif

// 16-byte global -> LDS copy. Async (ASYNCcnt, no VGPR bounce) when supported.
// Builtin signature (from clang diagnostic): (v4i AS1*, v4i AS3*, imm, imm).
__device__ __forceinline__ void cp_b128(const _Float16* src, _Float16* dst) {
#if ASYNC_LDS
  __builtin_amdgcn_global_load_async_to_lds_b128(
      (__attribute__((address_space(1))) v4i*)(src),
      (__attribute__((address_space(3))) v4i*)(dst), 0, 0);
#else
  *(v8h*)dst = *(const v8h*)src;
#endif
}

__device__ __forceinline__ void wait_async_cp() {
#if ASYNC_LDS
#if __has_builtin(__builtin_amdgcn_s_wait_asynccnt)
  __builtin_amdgcn_s_wait_asynccnt(0);
#else
  asm volatile("s_wait_asynccnt 0x0" ::: "memory");
#endif
#endif
}

// ---------------------------------------------------------------------------
// fp32 -> f16 elementwise convert (4 elems / thread)
// ---------------------------------------------------------------------------
__global__ __launch_bounds__(256) void f2h4_kernel(const float* __restrict__ in,
                                                   _Float16* __restrict__ out, int n) {
  int i = (blockIdx.x * 256 + threadIdx.x) * 4;
  if (i + 3 < n) {
    float4 v = *(const float4*)(in + i);
    out[i + 0] = (_Float16)v.x;
    out[i + 1] = (_Float16)v.y;
    out[i + 2] = (_Float16)v.z;
    out[i + 3] = (_Float16)v.w;
  }
}

// ---------------------------------------------------------------------------
// Tiled WMMA GEMM:  C[M,N] = alpha * A[M,K] @ B(K,N) + bias
//   Block = 256 threads (8 waves). C tile: 128x64. K-slab: 64.
//   Wave w computes rows [16w,16w+16) x 4 column tiles of 16.
//   LDS tiles are double-buffered and stored pre-swizzled in WMMA fragment
//   order, so each lane reads its v16h fragment with contiguous ds_load_b128.
//   A (and B when transB=1) stream via async global->LDS b128 copies that
//   overlap with the WMMAs of the previous slab.
// ---------------------------------------------------------------------------
__global__ __launch_bounds__(256) void gemm_wmma_f16(
    const _Float16* __restrict__ A, const _Float16* __restrict__ B,
    const float* __restrict__ bias, float* __restrict__ C,
    _Float16* __restrict__ Cbf,
    int M, int N, int K, int lda, int ldb, int ldc,
    long long strideA, long long strideB, long long strideC,
    int transB, int relu, float alpha) {
  // [buf][ (ks*8 + mtile) * 512 ]  : 128x64 A slab, fragment-swizzled
  __shared__ _Float16 sA[2][16 * 512];
  // [buf][ (ks*4 + ntile) * 512 ]  : 64x64 B slab, fragment-swizzled
  __shared__ _Float16 sB[2][8 * 512];

  const int tid  = threadIdx.x;
  const int wave = tid >> 5;
  const int lane = tid & 31;
  const int row0 = blockIdx.y * 128;
  const int col0 = blockIdx.x * 64;

  const _Float16* Ag = A + (size_t)blockIdx.z * strideA;
  const _Float16* Bg = B + (size_t)blockIdx.z * strideB;
  float*    Cp  = C   ? C   + (size_t)blockIdx.z * strideC : nullptr;
  _Float16* Chp = Cbf ? Cbf + (size_t)blockIdx.z * strideC : nullptr;

  v8f acc[4] = {};

  // --- stage A slab (128 x 64): 1024 chunks of 8 halves, 4 per thread ---
  // frag layout per 16x32 tile: lane = (m&15)+16*(kk>>4), idx=kk&15
  auto stageA = [&](int buf, int kt) {
    #pragma unroll
    for (int c = tid; c < 1024; c += 256) {
      int m   = c >> 3;             // 0..127
      int kk0 = (c & 7) << 3;       // 0..56
      const _Float16* src = Ag + (size_t)(row0 + m) * lda + kt + kk0;
      int ks  = kk0 >> 5;
      int kkl = kk0 & 31;
      int ln  = (m & 15) + ((kkl >> 4) << 4);
      int idx = kkl & 15;
      cp_b128(src, &sA[buf][(((ks << 3) + (m >> 4)) << 9) + ln * 16 + idx]);
    }
  };

  // --- stage B slab (64 x 64): frag layout lane=(n&15)+16*(kk>>4), idx=kk&15 ---
  auto stageB = [&](int buf, int kt) {
    if (transB) {
      // B element (k,n) = Bg[n*ldb + k]: contiguous along k -> async b128
      #pragma unroll
      for (int c = tid; c < 512; c += 256) {
        int n   = c >> 3;           // 0..63
        int kk0 = (c & 7) << 3;     // 0..56
        const _Float16* src = Bg + (size_t)(col0 + n) * ldb + kt + kk0;
        int ks  = kk0 >> 5;
        int kkl = kk0 & 31;
        int ln  = (n & 15) + ((kkl >> 4) << 4);
        cp_b128(src, &sB[buf][(((ks << 2) + (n >> 4)) << 9) + ln * 16 + (kkl & 15)]);
      }
    } else {
      // row-major KxN: contiguous along n -> lane scatter via ds stores
      #pragma unroll
      for (int c = tid; c < 512; c += 256) {
        int kk  = c >> 3;           // 0..63
        int n0c = (c & 7) << 3;     // 0..56
        v8h val = *(const v8h*)(Bg + (size_t)(kt + kk) * ldb + col0 + n0c);
        int ks  = kk >> 5;
        int kkl = kk & 31;
        int idx = kkl & 15;
        int lhi = (kkl >> 4) << 4;
        int tb  = ((ks << 2) + (n0c >> 4)) << 9;
        #pragma unroll
        for (int j = 0; j < 8; ++j) {
          int nn = (n0c & 15) + j;
          sB[buf][tb + (nn + lhi) * 16 + idx] = val[j];
        }
      }
    }
  };

  int buf = 0;
  stageA(0, 0);
  stageB(0, 0);
  wait_async_cp();
  __syncthreads();

  for (int kt = 0; kt < K; kt += TKS) {
    int nxt = kt + TKS;
    if (nxt < K) {
      if (nxt + TKS < K) {  // L2 prefetch two slabs ahead
        __builtin_prefetch(
            Ag + (size_t)(row0 + (tid >> 1)) * lda + nxt + TKS + ((tid & 1) << 5),
            0, 3);
      }
      stageA(buf ^ 1, nxt);   // async: overlaps with WMMAs below
      stageB(buf ^ 1, nxt);
    }

    #pragma unroll
    for (int ks = 0; ks < 2; ++ks) {
      v16h af = *(const v16h*)(&sA[buf][(((ks << 3) + wave) << 9) + lane * 16]);
      #pragma unroll
      for (int t = 0; t < 4; ++t) {
        v16h bf = *(const v16h*)(&sB[buf][(((ks << 2) + t) << 9) + lane * 16]);
        acc[t] = __builtin_amdgcn_wmma_f32_16x16x32_f16(
            false, af, false, bf, (short)0, acc[t], false, false);
      }
    }

    wait_async_cp();
    __syncthreads();
    buf ^= 1;
  }

  // --- epilogue: C/D layout: VGPR r -> M = r + 8*(lane>>4), N = lane&15 ---
  #pragma unroll
  for (int t = 0; t < 4; ++t) {
    int col = col0 + t * 16 + (lane & 15);
    float badd = bias ? bias[col] : 0.0f;
    #pragma unroll
    for (int r = 0; r < 8; ++r) {
      int rowr = row0 + wave * 16 + r + ((lane >> 4) << 3);
      float v = acc[t][r] * alpha + badd;
      if (relu) v = fmaxf(v, 0.0f);
      size_t ci = (size_t)rowr * ldc + col;
      if (Cp)  Cp[ci]  = v;
      if (Chp) Chp[ci] = (_Float16)v;
    }
  }
}

// ---------------------------------------------------------------------------
// Per-row exact top-k threshold (MSB radix select on order-preserving keys)
// fused with the reference's sparse softmax:
//   sparse = (s >= thr) ? s : 0 ;  attn = softmax(sparse)  (over ALL entries)
// One 256-thread block per score row (S = 2048).
// ---------------------------------------------------------------------------
__global__ __launch_bounds__(256) void topk_softmax_kernel(
    const float* __restrict__ scores, _Float16* __restrict__ attn,
    int S, int topk) {
  __shared__ unsigned keys[SEQ];
  __shared__ unsigned hist[256];
  __shared__ unsigned sel[2];
  __shared__ float red[256];

  const int row = blockIdx.x;
  const int tid = threadIdx.x;
  const float* sr = scores + (size_t)row * S;

  for (int i = tid; i < S; i += 256) {
    unsigned u = __float_as_uint(sr[i]);
    keys[i] = u ^ ((u >> 31) ? 0xFFFFFFFFu : 0x80000000u);
  }
  __syncthreads();

  unsigned prefix = 0, kneed = (unsigned)topk;
  for (int shift = 24; shift >= 0; shift -= 8) {
    hist[tid] = 0;
    __syncthreads();
    for (int i = tid; i < S; i += 256) {
      unsigned k = keys[i];
      bool ok = (shift == 24) || ((k >> (shift + 8)) == (prefix >> (shift + 8)));
      if (ok) atomicAdd(&hist[(k >> shift) & 255], 1u);
    }
    __syncthreads();
    if (tid == 0) {
      unsigned c = kneed, d = 0;
      for (int b = 255; b >= 0; --b) {
        unsigned h = hist[b];
        if (c <= h) { d = (unsigned)b; break; }
        c -= h;
      }
      sel[0] = d; sel[1] = c;
    }
    __syncthreads();
    prefix |= sel[0] << shift;
    kneed = sel[1];
    __syncthreads();
  }

  unsigned tu = (prefix & 0x80000000u) ? (prefix ^ 0x80000000u) : ~prefix;
  float thr = __uint_as_float(tu);

  // sparse values back into LDS (reuse keys as float storage), row max
  float* fk = (float*)keys;
  float lm = 0.0f;  // zeros are present (topk < S), so 0 is a valid identity
  for (int i = tid; i < S; i += 256) {
    unsigned k = keys[i];
    unsigned uu = (k & 0x80000000u) ? (k ^ 0x80000000u) : ~k;
    float f = __uint_as_float(uu);
    float sv = (f >= thr) ? f : 0.0f;
    lm = fmaxf(lm, sv);
    fk[i] = sv;
  }
  red[tid] = lm;
  __syncthreads();
  for (int off = 128; off > 0; off >>= 1) {
    if (tid < off) red[tid] = fmaxf(red[tid], red[tid + off]);
    __syncthreads();
  }
  float m = red[0];
  __syncthreads();

  float ls = 0.0f;
  for (int i = tid; i < S; i += 256) ls += __expf(fk[i] - m);
  red[tid] = ls;
  __syncthreads();
  for (int off = 128; off > 0; off >>= 1) {
    if (tid < off) red[tid] += red[tid + off];
    __syncthreads();
  }
  float inv = 1.0f / red[0];

  _Float16* ar = attn + (size_t)row * S;
  for (int i = tid; i < S; i += 256)
    ar[i] = (_Float16)(__expf(fk[i] - m) * inv);
}

// ---------------------------------------------------------------------------
// out = LayerNorm(a + b) * gamma + beta ; optional f16 copy for next GEMM.
// One 256-thread block per row of d=1024.
// ---------------------------------------------------------------------------
__global__ __launch_bounds__(256) void add_ln_kernel(
    const float* __restrict__ a, const float* __restrict__ b,
    const float* __restrict__ gamma, const float* __restrict__ beta,
    float* __restrict__ outf, _Float16* __restrict__ outh, int d) {
  __shared__ float sv[DMODEL];
  __shared__ float red[256];
  const int row = blockIdx.x;
  const int tid = threadIdx.x;
  const float* ar = a + (size_t)row * d;
  const float* br = b + (size_t)row * d;

  float s = 0.0f, s2 = 0.0f;
  for (int i = tid; i < d; i += 256) {
    float v = ar[i] + br[i];
    sv[i] = v;
    s += v;
    s2 += v * v;
  }
  red[tid] = s;
  __syncthreads();
  for (int off = 128; off > 0; off >>= 1) {
    if (tid < off) red[tid] += red[tid + off];
    __syncthreads();
  }
  float mu = red[0] / (float)d;
  __syncthreads();
  red[tid] = s2;
  __syncthreads();
  for (int off = 128; off > 0; off >>= 1) {
    if (tid < off) red[tid] += red[tid + off];
    __syncthreads();
  }
  float var = red[0] / (float)d - mu * mu;
  float rs = rsqrtf(var + LN_EPS);

  for (int i = tid; i < d; i += 256) {
    float o = (sv[i] - mu) * rs * gamma[i] + beta[i];
    size_t oi = (size_t)row * d + i;
    outf[oi] = o;
    if (outh) outh[oi] = (_Float16)o;
  }
}

// ---------------------------------------------------------------------------
// Host-side orchestration
// ---------------------------------------------------------------------------
extern "C" void kernel_launch(void* const* d_in, const int* in_sizes, int n_in,
                              void* d_out, int out_size, void* d_ws, size_t ws_size,
                              hipStream_t stream) {
  (void)in_sizes; (void)n_in; (void)out_size; (void)ws_size;

  const float* x      = (const float*)d_in[0];
  const float* Wq     = (const float*)d_in[1];
  const float* bq     = (const float*)d_in[2];
  const float* Wk     = (const float*)d_in[3];
  const float* bk     = (const float*)d_in[4];
  const float* Wv     = (const float*)d_in[5];
  const float* bv     = (const float*)d_in[6];
  const float* W1     = (const float*)d_in[7];
  const float* b1     = (const float*)d_in[8];
  const float* W2     = (const float*)d_in[9];
  const float* b2     = (const float*)d_in[10];
  const float* gamma1 = (const float*)d_in[11];
  const float* beta1  = (const float*)d_in[12];
  const float* gamma2 = (const float*)d_in[13];
  const float* beta2  = (const float*)d_in[14];
  float* out = (float*)d_out;

  // ---- workspace carve-up (256B aligned) ----
  char* w = (char*)d_ws;
  size_t off = 0;
  auto alloc = [&](size_t bytes) -> void* {
    void* p = w + off;
    off = (off + bytes + 255) & ~(size_t)255;
    return p;
  };
  _Float16* xb   = (_Float16*)alloc((size_t)BS * DMODEL * 2);
  _Float16* wqb  = (_Float16*)alloc((size_t)DMODEL * DMODEL * 2);
  _Float16* wkb  = (_Float16*)alloc((size_t)DMODEL * DMODEL * 2);
  _Float16* wvb  = (_Float16*)alloc((size_t)DMODEL * DMODEL * 2);
  _Float16* w1b  = (_Float16*)alloc((size_t)DMODEL * DFF * 2);
  _Float16* w2b  = (_Float16*)alloc((size_t)DFF * DMODEL * 2);
  _Float16* qb   = (_Float16*)alloc((size_t)BS * DMODEL * 2);
  _Float16* kb   = (_Float16*)alloc((size_t)BS * DMODEL * 2);
  _Float16* vb   = (_Float16*)alloc((size_t)BS * DMODEL * 2);
  float*    sc   = (float*)   alloc((size_t)BATCH * SEQ * SEQ * 4);
  _Float16* atb  = (_Float16*)alloc((size_t)BATCH * SEQ * SEQ * 2);
  float*    aout = (float*)   alloc((size_t)BS * DMODEL * 4);
  float*    out1 = (float*)   alloc((size_t)BS * DMODEL * 4);
  _Float16* o1b  = (_Float16*)alloc((size_t)BS * DMODEL * 2);
  _Float16* hb   = (_Float16*)alloc((size_t)BS * DFF * 2);
  float*    ffn  = (float*)   alloc((size_t)BS * DMODEL * 4);

  // ---- fp32 -> f16 operand conversions ----
  auto cvt = [&](const float* src, _Float16* dst, int n) {
    f2h4_kernel<<<n / 1024, 256, 0, stream>>>(src, dst, n);
  };
  cvt(x,  xb,  BS * DMODEL);
  cvt(Wq, wqb, DMODEL * DMODEL);
  cvt(Wk, wkb, DMODEL * DMODEL);
  cvt(Wv, wvb, DMODEL * DMODEL);
  cvt(W1, w1b, DMODEL * DFF);
  cvt(W2, w2b, DFF * DMODEL);

  // ---- QKV projections: [8192,1024] @ [1024,1024] + bias -> f16 ----
  {
    dim3 grid(DMODEL / 64, BS / 128, 1), block(256);
    gemm_wmma_f16<<<grid, block, 0, stream>>>(xb, wqb, bq, nullptr, qb,
        BS, DMODEL, DMODEL, DMODEL, DMODEL, DMODEL, 0, 0, 0, 0, 0, 1.0f);
    gemm_wmma_f16<<<grid, block, 0, stream>>>(xb, wkb, bk, nullptr, kb,
        BS, DMODEL, DMODEL, DMODEL, DMODEL, DMODEL, 0, 0, 0, 0, 0, 1.0f);
    gemm_wmma_f16<<<grid, block, 0, stream>>>(xb, wvb, bv, nullptr, vb,
        BS, DMODEL, DMODEL, DMODEL, DMODEL, DMODEL, 0, 0, 0, 0, 0, 1.0f);
  }

  // ---- scores = q @ k^T / sqrt(d), batched over B=4 ----
  {
    dim3 grid(SEQ / 64, SEQ / 128, BATCH), block(256);
    gemm_wmma_f16<<<grid, block, 0, stream>>>(qb, kb, nullptr, sc, nullptr,
        SEQ, SEQ, DMODEL, DMODEL, DMODEL, SEQ,
        (long long)SEQ * DMODEL, (long long)SEQ * DMODEL, (long long)SEQ * SEQ,
        /*transB=*/1, 0, 0.03125f /* 1/sqrt(1024) */);
  }

  // ---- exact top-k threshold + sparse softmax -> attn (f16) ----
  topk_softmax_kernel<<<BATCH * SEQ, 256, 0, stream>>>(sc, atb, SEQ, TOPK);

  // ---- attn_out = attn @ v, batched ----
  {
    dim3 grid(DMODEL / 64, SEQ / 128, BATCH), block(256);
    gemm_wmma_f16<<<grid, block, 0, stream>>>(atb, vb, nullptr, aout, nullptr,
        SEQ, DMODEL, SEQ, SEQ, DMODEL, DMODEL,
        (long long)SEQ * SEQ, (long long)SEQ * DMODEL, (long long)SEQ * DMODEL,
        0, 0, 1.0f);
  }

  // ---- out1 = LN(x + attn_out) (fp32 + f16 copy) ----
  add_ln_kernel<<<BS, 256, 0, stream>>>(x, aout, gamma1, beta1, out1, o1b, DMODEL);

  // ---- FFN: h = relu(out1 @ W1 + b1) -> f16 ; ffn = h @ W2 + b2 -> fp32 ----
  {
    dim3 grid1(DFF / 64, BS / 128, 1), block(256);
    gemm_wmma_f16<<<grid1, block, 0, stream>>>(o1b, w1b, b1, nullptr, hb,
        BS, DFF, DMODEL, DMODEL, DFF, DFF, 0, 0, 0, 0, /*relu=*/1, 1.0f);
    dim3 grid2(DMODEL / 64, BS / 128, 1);
    gemm_wmma_f16<<<grid2, block, 0, stream>>>(hb, w2b, b2, ffn, nullptr,
        BS, DMODEL, DFF, DFF, DMODEL, DMODEL, 0, 0, 0, 0, 0, 1.0f);
  }

  // ---- final: out = LN(out1 + ffn) ----
  add_ln_kernel<<<BS, 256, 0, stream>>>(out1, ffn, gamma2, beta2, out, nullptr, DMODEL);
}
